// MultiHeadAttention_16982300688411
// MI455X (gfx1250) — matrix operand order repaired and checked by reading, not compiled, not run on hardware
//
#include <hip/hip_runtime.h>

typedef __bf16 bf16;
typedef __attribute__((ext_vector_type(16))) __bf16 v16bf;
typedef __attribute__((ext_vector_type(8)))  __bf16 v8bf;
typedef __attribute__((ext_vector_type(8)))  float   v8f;
typedef __attribute__((ext_vector_type(4)))  int     v4i;

#define D_MODEL 1024
#define NHEAD   16
#define DK      64
#define BATCH   4
#define SEQ     2048
#define MTOT    (BATCH * SEQ)   // 8192

#ifndef __has_builtin
#define __has_builtin(x) 0
#endif
#if __has_builtin(__builtin_amdgcn_global_load_async_to_lds_b128) && \
    __has_builtin(__builtin_amdgcn_s_wait_asynccnt)
#define USE_ASYNC_COPY 1
#else
#define USE_ASYNC_COPY 0
#endif

// ---------------------------------------------------------------------------
// WMMA fragment helpers (16x16x32 bf16, wave32; contiguous k order chosen by this code)
// A 16x32 (MxK): lane = m + 16*h ; elems 0..15 = K[k0+16h .. +15], row m
// B 32x16 (KxN): lane = n + 16*h ; elems 0..15 = K[k0+16h .. +15], column n
// C/D 16x16 f32: vgpr j ; lanes 0-15 -> (row j, col lane), lanes 16-31 -> (row j+8, col lane-16)
// ---------------------------------------------------------------------------

__device__ __forceinline__ v16bf load_a_frag_f32(const float* __restrict__ row,
                                                 int k0, int h) {
  const float* p0 = row + k0 + 16 * h;
  const float* p1 = p0 + 8;
  v16bf r;
#pragma unroll
  for (int i = 0; i < 8; ++i) {
    r[i]     = (bf16)p0[i];
    r[8 + i] = (bf16)p1[i];
  }
  return r;
}

__device__ __forceinline__ v16bf load_a_frag_bf16(const bf16* __restrict__ row,
                                                  int k0, int h) {
  const v8bf a0 = *(const v8bf*)(row + k0 + 16 * h);
  const v8bf a1 = *(const v8bf*)(row + k0 + 16 * h + 8);
  v16bf r;
#pragma unroll
  for (int i = 0; i < 8; ++i) {
    r[i]     = a0[i];
    r[8 + i] = a1[i];
  }
  return r;
}

__device__ __forceinline__ v8f wmma_bf16(v16bf a, v16bf b, v8f c) {
  return __builtin_amdgcn_wmma_f32_16x16x32_bf16(false, a, false, b,
                                                 (short)0, c, false, false);
}

// ---------------------------------------------------------------------------
// fp32 -> bf16 elementwise conversion (weights), 8 elems/thread, vectorized.
// ---------------------------------------------------------------------------
__global__ __launch_bounds__(256) void cvt_f32_bf16_kernel(
    const float* __restrict__ in, bf16* __restrict__ out, int n) {
  const int i = (blockIdx.x * 256 + threadIdx.x) * 8;
  if (i >= n) return;
  v8bf r;
#pragma unroll
  for (int j = 0; j < 8; ++j) r[j] = (bf16)in[i + j];
  *(v8bf*)(out + i) = r;
}

// ---------------------------------------------------------------------------
// GEMM:  Out[M,N] = A[M,K] * Wb[N,K]^T      (y = x @ W.T), Wb already bf16.
// AMODE: 0 = A fp32 (convert on load), 1 = A bf16
// OMODE: 0 = bf16 row-major
//        1 = bf16 "V-transposed":  out[(b*1024 + n)*SEQ + s]  (b=m/SEQ, s=m%SEQ)
//        2 = fp32 row-major
// Block 128 = 4 waves.  Wave tile 32x64 (8 accums).  WG tile 128x64.
// Weight tile (64 x KCHUNK bf16, 16KB) staged in LDS via async global->LDS
// (ASYNCcnt path) when available.
// ---------------------------------------------------------------------------
#define KCHUNK 128
#define LDSB_STRIDE (KCHUNK + 16)  // pad 32B: rotate bank starting points

template <int AMODE, int OMODE>
__global__ __launch_bounds__(128) void gemm_wmma_kernel(
    const void* __restrict__ Ap, const bf16* __restrict__ Wb,
    void* __restrict__ Out, int M, int N, int K) {
  __shared__ alignas(32) bf16 ldsB[64][LDSB_STRIDE];

  const int ntile = blockIdx.x;              // 64-col tile
  const int mtile = blockIdx.y;              // 128-row tile
  const int wave  = threadIdx.x >> 5;
  const int lane  = threadIdx.x & 31;
  const int lm    = lane & 15;
  const int lh    = lane >> 4;

  const int row0 = mtile * 128 + wave * 32;  // wave owns rows row0..row0+31
  const int col0 = ntile * 64;

  v8f c[2][4] = {};

  const float* Af32  = (const float*)Ap;
  const bf16*  Abf16 = (const bf16*)Ap;

  for (int kb = 0; kb < K; kb += KCHUNK) {
    // ---- stage weight tile (rows col0..col0+63, k kb..kb+KCHUNK-1) -------
    {
      const int r     = threadIdx.x >> 1;          // 0..63
      const int halfk = (threadIdx.x & 1) * 64;    // 0 or 64 (elems)
      const bf16* src = Wb + (size_t)(col0 + r) * K + kb + halfk;
      bf16*       dst = &ldsB[r][halfk];
#if USE_ASYNC_COPY
#pragma unroll
      for (int i = 0; i < 8; ++i)
        __builtin_amdgcn_global_load_async_to_lds_b128(
            (v4i*)(src + i * 8), (v4i*)(dst + i * 8), 0, 0);
#else
#pragma unroll
      for (int i = 0; i < 8; ++i)
        *(v8bf*)(dst + i * 8) = *(const v8bf*)(src + i * 8);
#endif
    }
#if USE_ASYNC_COPY
    __builtin_amdgcn_s_wait_asynccnt(0);
#endif
    __syncthreads();

    // ---- compute over the staged K-chunk ---------------------------------
    for (int k0 = 0; k0 < KCHUNK; k0 += 32) {
      v16bf a[2];
#pragma unroll
      for (int sub = 0; sub < 2; ++sub) {
        const int arow = row0 + sub * 16 + lm;
        if (AMODE == 0)
          a[sub] = load_a_frag_f32(Af32 + (size_t)arow * K, kb + k0, lh);
        else
          a[sub] = load_a_frag_bf16(Abf16 + (size_t)arow * K, kb + k0, lh);
      }
#pragma unroll
      for (int t = 0; t < 4; ++t) {
        const v16bf b = *(const v16bf*)&ldsB[t * 16 + lm][k0 + 16 * lh];
        c[0][t] = wmma_bf16(a[0], b, c[0][t]);
        c[1][t] = wmma_bf16(a[1], b, c[1][t]);
      }
    }
    __syncthreads();
  }

  // ---- stores --------------------------------------------------------------
#pragma unroll
  for (int sub = 0; sub < 2; ++sub) {
    const int rbase = row0 + sub * 16;
    if (OMODE == 2) {
      float* out = (float*)Out;
#pragma unroll
      for (int t = 0; t < 4; ++t) {
        const int col = col0 + t * 16 + lm;
#pragma unroll
        for (int j = 0; j < 8; ++j)
          out[(size_t)(rbase + j + 8 * lh) * N + col] = c[sub][t][j];
      }
    } else if (OMODE == 0) {
      bf16* out = (bf16*)Out;
#pragma unroll
      for (int t = 0; t < 4; ++t) {
        const int col = col0 + t * 16 + lm;
#pragma unroll
        for (int j = 0; j < 8; ++j)
          out[(size_t)(rbase + j + 8 * lh) * N + col] = (bf16)c[sub][t][j];
      }
    } else {  // OMODE == 1 : per-head transposed V (contiguous along s)
      bf16*     out  = (bf16*)Out;
      const int bidx = rbase / SEQ;
      const int s0   = (rbase & (SEQ - 1)) + 8 * lh;
#pragma unroll
      for (int t = 0; t < 4; ++t) {
        const int n = col0 + t * 16 + lm;
        v8bf tmp;
#pragma unroll
        for (int j = 0; j < 8; ++j) tmp[j] = (bf16)c[sub][t][j];
        *(v8bf*)(out + ((size_t)bidx * D_MODEL + n) * SEQ + s0) = tmp;
      }
    }
  }
}

// ---------------------------------------------------------------------------
// Causal flash attention.
//   Q : bf16 [B*S, 1024] row-major    K : bf16 [B*S, 1024] row-major
//   Vt: bf16 [(b*1024 + h*64 + d), s] per-head transposed
//   O : bf16 [B*S, 1024] row-major
// ---------------------------------------------------------------------------
__global__ __launch_bounds__(128) void flash_attn_kernel(
    const bf16* __restrict__ Q, const bf16* __restrict__ Km,
    const bf16* __restrict__ Vt, bf16* __restrict__ O) {
  __shared__ alignas(16) bf16 ldsP[4][16][72];

  const int bh   = blockIdx.y;
  const int b    = bh >> 4;
  const int h    = bh & 15;
  const int wave = threadIdx.x >> 5;
  const int lane = threadIdx.x & 31;
  const int lm   = lane & 15;
  const int lh   = lane >> 4;

  const int q0 = blockIdx.x * 64 + wave * 16;

  const bf16* qrow = Q + ((size_t)(b * SEQ + q0 + lm)) * D_MODEL + h * DK;
  v16bf aq[2];
#pragma unroll
  for (int kk = 0; kk < 2; ++kk) aq[kk] = load_a_frag_bf16(qrow, kk * 32, lh);

  v8f   o[4] = {};
  float mrow[8], lrow[8];
#pragma unroll
  for (int j = 0; j < 8; ++j) {
    mrow[j] = -1e30f;
    lrow[j] = 0.0f;
  }

  const int kt_end = (q0 + 15) >> 6;
  for (int kt = 0; kt <= kt_end; ++kt) {
    const int n0 = kt * 64;

    v8f sacc[4] = {};
#pragma unroll
    for (int kk = 0; kk < 2; ++kk) {
#pragma unroll
      for (int t = 0; t < 4; ++t) {
        const bf16* kp = Km + (size_t)(b * SEQ + n0 + t * 16 + lm) * D_MODEL +
                         h * DK + kk * 32 + 16 * lh;
        sacc[t] = wmma_bf16(aq[kk], *(const v16bf*)kp, sacc[t]);
      }
    }

    float tmax[8];
#pragma unroll
    for (int j = 0; j < 8; ++j) {
      const int row = q0 + j + 8 * lh;
      float mx = -1e30f;
#pragma unroll
      for (int t = 0; t < 4; ++t) {
        const int col = n0 + t * 16 + lm;
        float v = sacc[t][j] * 0.125f;  // 1/sqrt(64)
        if (col > row) v = -1e30f;
        sacc[t][j] = v;
        mx = fmaxf(mx, v);
      }
      tmax[j] = mx;
    }
#pragma unroll
    for (int off = 1; off < 16; off <<= 1)
#pragma unroll
      for (int j = 0; j < 8; ++j)
        tmax[j] = fmaxf(tmax[j], __shfl_xor(tmax[j], off, 32));

    float alpha[8], rsum[8];
#pragma unroll
    for (int j = 0; j < 8; ++j) {
      const float mn = fmaxf(mrow[j], tmax[j]);
      alpha[j]       = __expf(mrow[j] - mn);
      mrow[j]        = mn;
      float s        = 0.0f;
#pragma unroll
      for (int t = 0; t < 4; ++t) {
        const float p = __expf(sacc[t][j] - mn);
        sacc[t][j]    = p;
        s += p;
      }
      rsum[j] = s;
    }
#pragma unroll
    for (int off = 1; off < 16; off <<= 1)
#pragma unroll
      for (int j = 0; j < 8; ++j) rsum[j] += __shfl_xor(rsum[j], off, 32);
#pragma unroll
    for (int j = 0; j < 8; ++j) lrow[j] = lrow[j] * alpha[j] + rsum[j];
#pragma unroll
    for (int t = 0; t < 4; ++t)
#pragma unroll
      for (int j = 0; j < 8; ++j) o[t][j] *= alpha[j];

#pragma unroll
    for (int t = 0; t < 4; ++t)
#pragma unroll
      for (int j = 0; j < 8; ++j)
        ldsP[wave][j + 8 * lh][t * 16 + lm] = (bf16)sacc[t][j];

#pragma unroll
    for (int kk = 0; kk < 2; ++kk) {
      const bf16* pr = &ldsP[wave][lm][kk * 32 + 16 * lh];
      const v8bf  p0 = *(const v8bf*)pr;
      const v8bf  p1 = *(const v8bf*)(pr + 8);
      v16bf ap;
#pragma unroll
      for (int i = 0; i < 8; ++i) {
        ap[i]     = p0[i];
        ap[8 + i] = p1[i];
      }
#pragma unroll
      for (int t = 0; t < 4; ++t) {
        const bf16* vp = Vt +
                         ((size_t)b * D_MODEL + h * DK + t * 16 + lm) * SEQ +
                         n0 + kk * 32 + 16 * lh;
        o[t] = wmma_bf16(ap, *(const v16bf*)vp, o[t]);
      }
    }
  }

#pragma unroll
  for (int t = 0; t < 4; ++t) {
    const int col = h * DK + t * 16 + lm;
#pragma unroll
    for (int j = 0; j < 8; ++j) {
      const int row = q0 + j + 8 * lh;
      O[(size_t)(b * SEQ + row) * D_MODEL + col] = (bf16)(o[t][j] / lrow[j]);
    }
  }
}

// ---------------------------------------------------------------------------
extern "C" void kernel_launch(void* const* d_in, const int* in_sizes, int n_in,
                              void* d_out, int out_size, void* d_ws,
                              size_t ws_size, hipStream_t stream) {
  (void)in_sizes; (void)n_in; (void)out_size; (void)ws_size;
  const float* query = (const float*)d_in[0];
  const float* key   = (const float*)d_in[1];
  const float* value = (const float*)d_in[2];
  // d_in[3]: causal mask (implicit in the flash kernel) -- unused
  const float* w_q = (const float*)d_in[4];
  const float* w_k = (const float*)d_in[5];
  const float* w_v = (const float*)d_in[6];
  const float* w_o = (const float*)d_in[7];
  float*       out = (float*)d_out;

  const size_t elems  = (size_t)MTOT * D_MODEL;      // 8M
  const int    welems = D_MODEL * D_MODEL;           // 1M per weight
  bf16* qb  = (bf16*)d_ws;
  bf16* kb  = qb + elems;
  bf16* vt  = kb + elems;
  bf16* ob  = vt + elems;
  bf16* wqb = ob + elems;
  bf16* wkb = wqb + welems;
  bf16* wvb = wkb + welems;
  bf16* wob = wvb + welems;   // total: 64 MB + 8 MB workspace

  // 1) convert weights to bf16 once (removes per-tile cvt VALU from GEMMs)
  {
    dim3 cg(welems / (256 * 8)), cb(256);
    cvt_f32_bf16_kernel<<<cg, cb, 0, stream>>>(w_q, wqb, welems);
    cvt_f32_bf16_kernel<<<cg, cb, 0, stream>>>(w_k, wkb, welems);
    cvt_f32_bf16_kernel<<<cg, cb, 0, stream>>>(w_v, wvb, welems);
    cvt_f32_bf16_kernel<<<cg, cb, 0, stream>>>(w_o, wob, welems);
  }

  dim3 gg(D_MODEL / 64, MTOT / 128);  // (16, 64)
  dim3 gb(128);

  gemm_wmma_kernel<0, 0><<<gg, gb, 0, stream>>>(query, wqb, qb, MTOT, D_MODEL, D_MODEL);
  gemm_wmma_kernel<0, 0><<<gg, gb, 0, stream>>>(key,   wkb, kb, MTOT, D_MODEL, D_MODEL);
  gemm_wmma_kernel<0, 1><<<gg, gb, 0, stream>>>(value, wvb, vt, MTOT, D_MODEL, D_MODEL);

  dim3 ga(SEQ / 64, BATCH * NHEAD);  // (32, 64)
  flash_attn_kernel<<<ga, gb, 0, stream>>>(qb, kb, vt, ob);

  gemm_wmma_kernel<1, 2><<<gg, gb, 0, stream>>>(ob, wob, out, MTOT, D_MODEL, D_MODEL);
}
